// Head_84662395338853
// MI455X (gfx1250) — matrix operand order repaired and checked by reading, not compiled, not run on hardware
//
#include <hip/hip_runtime.h>

// ---------------------------------------------------------------------------
// Causal single-head attention for MI455X (gfx1250, wave32, WMMA).
//   Q = x Wq, K = x Wk, V = x Wv   (f16 WMMA, f32 accumulate)
//   out = softmax(mask(QK^T / sqrt(H))) V
// B=128, T=256, C=1024, H=128
// ---------------------------------------------------------------------------

typedef __attribute__((ext_vector_type(16))) _Float16 v16h;
typedef __attribute__((ext_vector_type(8)))  _Float16 v8h;
typedef __attribute__((ext_vector_type(8)))  float    v8f;
typedef __attribute__((ext_vector_type(4)))  int      v4i;

static constexpr int B_ = 128;
static constexpr int T_ = 256;
static constexpr int C_ = 1024;
static constexpr int H_ = 128;
static constexpr float kScale = 0.08838834764831845f; // 1/sqrt(128)

#if __has_builtin(__builtin_amdgcn_global_load_async_to_lds_b128)
#define HAVE_ASYNC_LDS 1
// async builtin takes (int4 AS1*, int4 AS3*, imm offset, imm cpol)
typedef __attribute__((address_space(1))) v4i* gp_v4i;
typedef __attribute__((address_space(3))) v4i* lp_v4i;
#else
#define HAVE_ASYNC_LDS 0
#endif

// Scheduling pins: 0x008 = MFMA/WMMA, 0x020 = VMEM read, 0x100 = DS read.
#if __has_builtin(__builtin_amdgcn_sched_group_barrier)
#define SGB(mask, cnt) __builtin_amdgcn_sched_group_barrier((mask), (cnt), 0)
#else
#define SGB(mask, cnt)
#endif

static __device__ __forceinline__ v16h cat8(v8h lo, v8h hi) {
  return __builtin_shufflevector(lo, hi, 0, 1, 2, 3, 4, 5, 6, 7,
                                         8, 9, 10, 11, 12, 13, 14, 15);
}

static __device__ __forceinline__ v8h cvt8(v8f v) {
  v8h r;
#pragma unroll
  for (int i = 0; i < 8; ++i) r[i] = (_Float16)v[i];
  return r;
}

static __device__ __forceinline__ v8f wmma16(v16h a, v16h b, v8f c) {
  // D = A(16x32 f16) * B(32x16 f16) + C(16x16 f32)
  return __builtin_amdgcn_wmma_f32_16x16x32_f16(
      /*neg_a=*/false, a, /*neg_b=*/false, b,
      /*c_mod=*/(short)0, c, /*reuse_a=*/false, /*reuse_b=*/false);
}

// ---------------------------------------------------------------------------
// Kernel 1: Wq/Wk/Wv (C x H, f32) -> Wt (3 x H x C, f16, transposed)
// so WMMA B-fragments (fixed column, contiguous K) are contiguous loads.
// ---------------------------------------------------------------------------
__global__ void cvt_w(const float* __restrict__ Wq, const float* __restrict__ Wk,
                      const float* __restrict__ Wv, _Float16* __restrict__ Wt,
                      int n) {
  int i = blockIdx.x * blockDim.x + threadIdx.x;
  if (i >= n) return;
  int mat = i / (C_ * H_);
  int rem = i - mat * (C_ * H_);
  int c = rem / H_;
  int h = rem - c * H_;
  const float* W = (mat == 0) ? Wq : ((mat == 1) ? Wk : Wv);
  Wt[(size_t)mat * H_ * C_ + (size_t)h * C_ + c] = (_Float16)W[(size_t)c * H_ + h];
}

// ---------------------------------------------------------------------------
// Kernel 2: QKV projection GEMM with fused f32->f16 conversion of x.
// grid.x = 3 * 256 blocks; block = 256 threads = 8 waves.
// W K-chunks are staged in double-buffered LDS via async global->LDS DMA
// (ASYNCcnt path) and shared by all 8 waves; A fragments are prefetched one
// chunk ahead in registers. The loop body is branch-free (next index clamped)
// and sched_group_barrier pins "16 DS reads, then 8 WMMAs" so the matrix ops
// ride partial DScnt waits instead of one full wait per WMMA.
// mat 0 -> Q (B*T x H row-major f16), mat 1 -> K (same), mat 2 -> V^T (B,H,T).
// ---------------------------------------------------------------------------
static constexpr int WSTR = 40;  // LDS row stride in halves (80B, 16B-aligned)

__global__ __launch_bounds__(256) void qkv_gemm(
    const float* __restrict__ x, const _Float16* __restrict__ Wt,
    _Float16* __restrict__ Qh, _Float16* __restrict__ Kh,
    _Float16* __restrict__ Vth) {
  __shared__ _Float16 sW[2][H_ * WSTR];   // 2 x 10 KB

  const int mat   = blockIdx.x >> 8;   // 0..2
  const int mb    = blockIdx.x & 255;  // M block (128 rows each)
  const int wave  = threadIdx.x >> 5;
  const int lane  = threadIdx.x & 31;
  const int lidx  = lane & 15;
  const int lhalf = lane >> 4;
  const int koff  = lhalf * 8;
  const int row0  = mb * 128 + wave * 16;
  const int tid   = threadIdx.x;
  const _Float16* W = Wt + (size_t)mat * (H_ * C_);

  // ---- staging: 512 16-byte segments per 32x128 W chunk, 2 per thread ----
  auto stage = [&](int c, int buf) {
#pragma unroll
    for (int i = 0; i < 2; ++i) {
      const int s = tid * 2 + i;           // 0..511
      const int h = s >> 2, q = s & 3;
      const _Float16* g = W + (size_t)h * C_ + c * 32 + q * 8;
      _Float16* l = &sW[buf][h * WSTR + q * 8];
#if HAVE_ASYNC_LDS
      __builtin_amdgcn_global_load_async_to_lds_b128(
          (gp_v4i)(void*)g, (lp_v4i)(void*)l, /*offset=*/0, /*cpol=*/0);
#else
      *(v8h*)l = *(const v8h*)g;
#endif
    }
  };
  auto wait_stage = [&]() {
#if HAVE_ASYNC_LDS
#if __has_builtin(__builtin_amdgcn_s_wait_asynccnt)
    __builtin_amdgcn_s_wait_asynccnt(0);
#else
    asm volatile("s_wait_asynccnt 0x0" ::: "memory");
#endif
#endif
    __syncthreads();
  };

  v8f zero = {};
  v8f acc[8];
#pragma unroll
  for (int i = 0; i < 8; ++i) acc[i] = zero;

  // A-fragment source: x rows in f32, converted in-register.
  const float* arow = x + (size_t)(row0 + lidx) * C_;

  stage(0, 0);
  // prefetch A chunk 0
  v8f a_lo = *(const v8f*)(arow + koff);
  v8f a_hi = *(const v8f*)(arow + koff + 16);
  wait_stage();

  for (int c = 0; c < C_ / 32; ++c) {
    const int buf = c & 1;
    const int cn  = (c + 1 < C_ / 32) ? (c + 1) : c;  // clamp: branch-free body
    v16h a = cat8(cvt8(a_lo), cvt8(a_hi));
    stage(cn, buf ^ 1);                          // async DMA next W chunk
    a_lo = *(const v8f*)(arow + cn * 32 + koff); // prefetch next A
    a_hi = *(const v8f*)(arow + cn * 32 + koff + 16);

    // Preload all 8 B fragments, then fire 8 independent WMMAs.
    v16h bfrag[8];
#pragma unroll
    for (int nt = 0; nt < 8; ++nt) {
      const _Float16* wp = &sW[buf][(nt * 16 + lidx) * WSTR + lhalf * 8];
      bfrag[nt] = cat8(*(const v8h*)wp, *(const v8h*)(wp + 16));
    }
#pragma unroll
    for (int nt = 0; nt < 8; ++nt) acc[nt] = wmma16(a, bfrag[nt], acc[nt]);

    SGB(0x100, 16);  // all 16 ds_load_b128 first
    SGB(0x008, 8);   // then 8 back-to-back WMMAs (partial DScnt waits)

    wait_stage();    // next chunk staged for everyone; all reads of buf done
  }

#pragma unroll
  for (int nt = 0; nt < 8; ++nt) {
    const int n = nt * 16 + lidx;
#pragma unroll
    for (int r = 0; r < 8; ++r) {
      const int m = r + 8 * lhalf;        // C/D layout: VGPR r, half-wave lhalf
      const int g = row0 + m;
      const _Float16 hv = (_Float16)acc[nt][r];
      if (mat == 0) {
        Qh[(size_t)g * H_ + n] = hv;
      } else if (mat == 1) {
        Kh[(size_t)g * H_ + n] = hv;
      } else {
        const int bb = g >> 8, tt = g & 255;
        Vth[((size_t)bb * H_ + n) * T_ + tt] = hv;  // transposed V
      }
    }
  }
}

// ---------------------------------------------------------------------------
// Kernel 3: causal attention. block = 128 threads = 4 waves.
// One wave owns one (batch, 16-query tile): scores via WMMA, f32 softmax with
// cross-lane reductions, probabilities staged in LDS (cross-lane transpose to
// A-fragment layout), then S@V^T via WMMA.
// ---------------------------------------------------------------------------
__global__ __launch_bounds__(128) void attn(
    const _Float16* __restrict__ Qh, const _Float16* __restrict__ Kh,
    const _Float16* __restrict__ Vth, float* __restrict__ out) {
  constexpr int SSTR = 264;                 // row stride (halves), 528B = 33*16B
  __shared__ _Float16 sS[4][16 * SSTR];

  const int wave  = threadIdx.x >> 5;
  const int lane  = threadIdx.x & 31;
  const int lidx  = lane & 15;
  const int lhalf = lane >> 4;
  const int koff  = lhalf * 8;
  const int gw    = blockIdx.x * 4 + wave;
  const int b     = gw >> 4;                // batch
  const int qt    = gw & 15;                // query tile (16 rows)
  _Float16* S = sS[wave];

  // ---- Q A-fragments for all 4 K-chunks of H=128 ----
  v16h aq[4];
  const size_t qrow = ((size_t)b * T_ + qt * 16 + lidx) * H_;
#pragma unroll
  for (int c = 0; c < 4; ++c) {
    v8h lo = *(const v8h*)(Qh + qrow + c * 32 + koff);
    v8h hi = *(const v8h*)(Qh + qrow + c * 32 + koff + 16);
    aq[c] = cat8(lo, hi);
  }

  // ---- Phase 1: scores = Q K^T * scale, causal mask ----
  float sc[16][8];
#pragma unroll
  for (int kt = 0; kt < 16; ++kt) {
    if (kt <= qt) {
      // Preload the 4 K B-fragments for this key tile, then 4 WMMAs.
      const _Float16* krow =
          Kh + ((size_t)b * T_ + kt * 16 + lidx) * H_ + lhalf * 8;
      v16h bk[4];
#pragma unroll
      for (int c = 0; c < 4; ++c)
        bk[c] = cat8(*(const v8h*)(krow + c * 32),
                     *(const v8h*)(krow + c * 32 + 16));
      v8f acc = {};
#pragma unroll
      for (int c = 0; c < 4; ++c) acc = wmma16(aq[c], bk[c], acc);
      SGB(0x020, 8);  // 8 global b128 loads first
      SGB(0x008, 4);  // then 4 WMMAs
#pragma unroll
      for (int r = 0; r < 8; ++r) {
        const int m = r + 8 * lhalf;
        float v = acc[r] * kScale;
        if (kt == qt && lidx > m) v = -3.0e38f;  // causal mask on diagonal tile
        sc[kt][r] = v;
      }
    } else {
#pragma unroll
      for (int r = 0; r < 8; ++r) sc[kt][r] = -3.0e38f;
    }
  }

  // ---- Phase 2: softmax per row (row m lives across 16 lanes of one half) --
#pragma unroll
  for (int r = 0; r < 8; ++r) {
    const int m = r + 8 * lhalf;
    float mx = -3.0e38f;
#pragma unroll
    for (int kt = 0; kt < 16; ++kt) mx = fmaxf(mx, sc[kt][r]);
    mx = fmaxf(mx, __shfl_xor(mx, 1, 32));
    mx = fmaxf(mx, __shfl_xor(mx, 2, 32));
    mx = fmaxf(mx, __shfl_xor(mx, 4, 32));
    mx = fmaxf(mx, __shfl_xor(mx, 8, 32));
    float sum = 0.0f;
#pragma unroll
    for (int kt = 0; kt < 16; ++kt) {
      float p = __expf(sc[kt][r] - mx);
      sc[kt][r] = p;
      sum += p;
    }
    sum += __shfl_xor(sum, 1, 32);
    sum += __shfl_xor(sum, 2, 32);
    sum += __shfl_xor(sum, 4, 32);
    sum += __shfl_xor(sum, 8, 32);
    const float inv = 1.0f / sum;
#pragma unroll
    for (int kt = 0; kt < 16; ++kt)
      S[m * SSTR + kt * 16 + lidx] = (_Float16)(sc[kt][r] * inv);
  }
  // Same-wave LDS ops are in-order; DScnt waits are inserted by the compiler.

  // ---- Phase 3: out = S @ V (V stored transposed: (B,H,T)) ----
  v8f zero = {};
  v8f o[8];
#pragma unroll
  for (int nt = 0; nt < 8; ++nt) o[nt] = zero;

#pragma unroll
  for (int cc = 0; cc < 8; ++cc) {          // K2 = 256 keys in chunks of 32
    if (2 * cc <= qt) {                     // fully-masked chunks contribute 0
      const _Float16* sp = S + lidx * SSTR + cc * 32 + koff;
      v16h as = cat8(*(const v8h*)sp, *(const v8h*)(sp + 16));
      // Preload all 8 V B-fragments, then 8 independent WMMAs.
      v16h bv[8];
#pragma unroll
      for (int nt = 0; nt < 8; ++nt) {
        const _Float16* vp =
            Vth + ((size_t)b * H_ + nt * 16 + lidx) * T_ + cc * 32 + lhalf * 8;
        bv[nt] = cat8(*(const v8h*)vp, *(const v8h*)(vp + 16));
      }
#pragma unroll
      for (int nt = 0; nt < 8; ++nt) o[nt] = wmma16(as, bv[nt], o[nt]);
      SGB(0x100, 2);   // 2 ds_load_b128 (S A-fragment)
      SGB(0x020, 16);  // 16 global b128 loads (V fragments)
      SGB(0x008, 8);   // then 8 WMMAs
    }
  }

#pragma unroll
  for (int nt = 0; nt < 8; ++nt) {
    const int n = nt * 16 + lidx;
#pragma unroll
    for (int r = 0; r < 8; ++r) {
      const int m = r + 8 * lhalf;
      out[((size_t)b * T_ + qt * 16 + m) * H_ + n] = o[nt][r];
    }
  }
}

// ---------------------------------------------------------------------------
extern "C" void kernel_launch(void* const* d_in, const int* in_sizes, int n_in,
                              void* d_out, int out_size, void* d_ws,
                              size_t ws_size, hipStream_t stream) {
  (void)in_sizes; (void)n_in; (void)out_size; (void)ws_size;
  const float* x  = (const float*)d_in[0];
  const float* Wq = (const float*)d_in[1];
  const float* Wk = (const float*)d_in[2];
  const float* Wv = (const float*)d_in[3];
  float* out = (float*)d_out;

  // Workspace layout (bytes):
  //   Wt  : 0      .. 768 KB   (3*H*C f16, transposed)
  //   Qh  : 1 MB   .. +8 MB    (f16)
  //   Kh  : 9 MB   .. +8 MB    (f16)
  //   Vth : 17 MB  .. +8 MB    (f16, transposed (B,H,T))
  char* ws = (char*)d_ws;
  _Float16* Wt  = (_Float16*)(ws);
  _Float16* Qh  = (_Float16*)(ws + (size_t)1048576);
  _Float16* Kh  = (_Float16*)(ws + (size_t)9437184);
  _Float16* Vth = (_Float16*)(ws + (size_t)17825792);

  const int nw = 3 * C_ * H_;                              // 393,216
  cvt_w<<<nw / 256, 256, 0, stream>>>(Wq, Wk, Wv, Wt, nw);
  qkv_gemm<<<3 * 256, 256, 0, stream>>>(x, Wt, Qh, Kh, Vth);
  attn<<<(B_ * (T_ / 16)) / 4, 128, 0, stream>>>(Qh, Kh, Vth, out);
}
